// MemoryAsContextTransformer_16183436771577
// MI455X (gfx1250) — matrix operand order repaired and checked by reading, not compiled, not run on hardware
//
#include <hip/hip_runtime.h>
#include <hip/hip_bf16.h>

// ---------------------------------------------------------------------------
// MemoryAsContext transformer block for gfx1250 (MI455X), f16 WMMA pipeline.
// ---------------------------------------------------------------------------

typedef __attribute__((ext_vector_type(16))) _Float16 v16h;
typedef __attribute__((ext_vector_type(8)))  _Float16 v8h;
typedef __attribute__((ext_vector_type(8)))  float    v8f;

#define NB   2
#define NN   4096
#define ND   1024
#define NH   16
#define NDH  64
#define NSEG 512
#define NPM  16
#define NQKV 3072
#define NROWS (NB * NN)   // 8192

// ---------------------------------------------------------------------------
// WMMA helpers (ISA-documented VGPR fragment layouts, wave32)
// ---------------------------------------------------------------------------
__device__ __forceinline__ v8f wmma_f16(v16h a, v16h b, v8f c) {
  // D = A(16x32 f16) x B(32x16 f16) + C(16x16 f32)
  return __builtin_amdgcn_wmma_f32_16x16x32_f16(false, a, false, b, (short)0, c,
                                                false, false);
}

// A fragment: lane L -> row m0+(L&15); halves [k0+8*(L>>4) .. +7] and [+16 .. +23]
__device__ __forceinline__ v16h load_a_frag(const _Float16* p, int ld, int m0,
                                            int k0, int lane) {
  int m  = m0 + (lane & 15);
  int kk = k0 + 8 * (lane >> 4);
  const _Float16* base = p + (size_t)m * ld + kk;
  v8h lo = *(const v8h*)(base);
  v8h hi = *(const v8h*)(base + 16);
  v16h r;
#pragma unroll
  for (int i = 0; i < 8; ++i) { r[i] = lo[i]; r[i + 8] = hi[i]; }
  return r;
}

// B fragment: lane L -> column n0+(L&15); rows [k0+8*(L>>4) .. +7] and [+16 .. +23]
__device__ __forceinline__ v16h load_b_frag(const _Float16* p, int ld, int k0,
                                            int n0, int lane) {
  int n  = n0 + (lane & 15);
  int kb = k0 + 8 * (lane >> 4);
  v16h r;
#pragma unroll
  for (int i = 0; i < 8; ++i) {
    r[i]     = p[(size_t)(kb + i) * ld + n];
    r[i + 8] = p[(size_t)(kb + 16 + i) * ld + n];
  }
  return r;
}

// ---------------------------------------------------------------------------
// Small conversion kernels (f32 -> f16 staging; L2-resident on MI455X)
// ---------------------------------------------------------------------------
__global__ void cvt_f32_to_f16(const float* __restrict__ in,
                               _Float16* __restrict__ out, int n) {
  int i = blockIdx.x * 256 + threadIdx.x;
  if (i < n) out[i] = (_Float16)in[i];
}

// pmem [2][H][PM][DH] f32 -> pmkt [H][DH][PM] f16 (transposed K), pmv [H][PM][DH] f16
__global__ void cvt_pmem(const float* __restrict__ pmem,
                         _Float16* __restrict__ pmkt,
                         _Float16* __restrict__ pmv) {
  int i = blockIdx.x * 256 + threadIdx.x;  // over NH*NPM*NDH = 16384
  if (i >= NH * NPM * NDH) return;
  int dh = i & 63;
  int p  = (i >> 6) & 15;
  int h  = i >> 10;
  float kval = pmem[(((size_t)0 * NH + h) * NPM + p) * NDH + dh];
  float vval = pmem[(((size_t)1 * NH + h) * NPM + p) * NDH + dh];
  pmkt[((size_t)h * NDH + dh) * NPM + p] = (_Float16)kval;
  pmv [((size_t)h * NPM + p) * NDH + dh] = (_Float16)vval;
}

// RoPE angle tables: cos/sin[t][dh], repeated-pair layout (ang[2i]==ang[2i+1]).
// 2 x 1 MB f32 -> stays L2 resident; removes per-element transcendental
// expansion (powf+sincos ~60 VALU ops each) from the QKV GEMM epilogue.
__global__ void rope_table_kernel(float* __restrict__ cosT,
                                  float* __restrict__ sinT) {
  int i  = blockIdx.x * 256 + threadIdx.x;   // over NN*NDH
  int dh = i & 63;
  int t  = i >> 6;
  float fpair = (float)(dh & ~1);
  float ang = (float)t * __powf(10000.0f, -fpair * (1.0f / (float)NDH));
  cosT[i] = __cosf(ang);
  sinT[i] = __sinf(ang);
}

// ---------------------------------------------------------------------------
// RMSNorm: one block per row (D=1024), f32 math, f16 output
// ---------------------------------------------------------------------------
__global__ void rmsnorm_kernel(const float* __restrict__ x,
                               const float* __restrict__ gamma,
                               _Float16* __restrict__ xn) {
  __shared__ float red[256];
  int row = blockIdx.x;
  const float* xr = x + (size_t)row * ND;
  float ss = 0.f;
  for (int i = threadIdx.x; i < ND; i += 256) { float v = xr[i]; ss += v * v; }
  red[threadIdx.x] = ss;
  __syncthreads();
  for (int s = 128; s > 0; s >>= 1) {
    if ((int)threadIdx.x < s) red[threadIdx.x] += red[threadIdx.x + s];
    __syncthreads();
  }
  float inv = rsqrtf(red[0] * (1.0f / (float)ND) + 1e-6f);
  for (int i = threadIdx.x; i < ND; i += 256)
    xn[(size_t)row * ND + i] = (_Float16)(xr[i] * inv * gamma[i]);
}

// ---------------------------------------------------------------------------
// QKV GEMM (8192x1024 @ 1024x3072) with fused RoPE + scatter epilogue.
// Wave tile: 32(M) x 64(N) = 2x4 WMMA tiles. 8 waves / block.
// ---------------------------------------------------------------------------
__global__ void qkv_gemm_kernel(const _Float16* __restrict__ Xn,
                                const _Float16* __restrict__ Wq,
                                const float* __restrict__ cosT,
                                const float* __restrict__ sinT,
                                _Float16* __restrict__ Qb,
                                _Float16* __restrict__ KTb,
                                _Float16* __restrict__ Vb) {
  const int K = ND, Nout = NQKV;
  int lane = threadIdx.x & 31;
  int wid  = blockIdx.x * 8 + (threadIdx.x >> 5);
  const int WN = Nout / 64;           // 48 wave-columns
  int wm = wid / WN, wn = wid % WN;
  int m0 = wm * 32, n0 = wn * 64;

  v8f acc[2][4];
#pragma unroll
  for (int i = 0; i < 2; ++i)
#pragma unroll
    for (int j = 0; j < 4; ++j) acc[i][j] = (v8f){};

  for (int k = 0; k < K; k += 32) {
    v16h a0 = load_a_frag(Xn, K, m0, k, lane);
    v16h a1 = load_a_frag(Xn, K, m0 + 16, k, lane);
    v16h bf[4];
#pragma unroll
    for (int j = 0; j < 4; ++j) bf[j] = load_b_frag(Wq, Nout, k, n0 + 16 * j, lane);
#pragma unroll
    for (int j = 0; j < 4; ++j) {
      acc[0][j] = wmma_f16(a0, bf[j], acc[0][j]);
      acc[1][j] = wmma_f16(a1, bf[j], acc[1][j]);
    }
  }

  // Epilogue: RoPE (q,k) via lane-pair shuffle + table lookup, scatter.
#pragma unroll
  for (int i = 0; i < 2; ++i) {
#pragma unroll
    for (int j = 0; j < 4; ++j) {
      int nt  = n0 + 16 * j;
      int sec = nt >> 10;          // 0=q, 1=k, 2=v (uniform per tile)
      int rem = nt & 1023;
      int h   = rem >> 6;
      int dh  = (rem & 63) + (lane & 15);
#pragma unroll
      for (int r = 0; r < 8; ++r) {
        int row = m0 + 16 * i + r + 8 * (lane >> 4);
        int b   = row >> 12;
        int t   = row & (NN - 1);
        float val = acc[i][j][r];
        if (sec == 2) {
          Vb[(((size_t)(b * NH + h)) * NN + t) * NDH + dh] = (_Float16)val;
        } else {
          float part = __shfl_xor(val, 1, 32);      // rotation partner (dh^1)
          int   ti = t * NDH + dh;
          float cs = cosT[ti], sn = sinT[ti];
          float out = (dh & 1) ? (val * cs + part * sn)
                               : (val * cs - part * sn);
          if (sec == 0)
            Qb[(((size_t)(b * NH + h)) * NN + t) * NDH + dh] = (_Float16)out;
          else
            KTb[(((size_t)(b * NH + h)) * NDH + dh) * NN + t] = (_Float16)out;
        }
      }
    }
  }
}

// ---------------------------------------------------------------------------
// Segmented causal attention with persistent memory, flash-style online
// softmax. One wave owns 16 query rows; keys processed 32 at a time
// (2 S-tiles -> P 16x32 via LDS transpose -> 4 PV WMMAs over dh).
// ---------------------------------------------------------------------------
__global__ void attn_kernel(const _Float16* __restrict__ Qb,
                            const _Float16* __restrict__ KTb,
                            const _Float16* __restrict__ Vb,
                            const _Float16* __restrict__ PMKT,
                            const _Float16* __restrict__ PMV,
                            _Float16* __restrict__ AttO) {
  __shared__ _Float16 Plds[8][16][40];   // per-wave 16x32 P tile (+pad)

  int lane = threadIdx.x & 31;
  int wave = threadIdx.x >> 5;
  int wid  = blockIdx.x * 8 + wave;      // 0..8191
  int q16  = wid & 255;                  // 256 query-16 blocks per (b,h)
  int bh   = wid >> 8;
  int h    = bh & 15;
  int b    = bh >> 4;
  int t0   = q16 * 16;
  int w    = t0 / NSEG;
  int qseg = t0 % NSEG;

  const _Float16* Qp  = Qb  + ((size_t)(b * NH + h)) * NN * NDH;
  const _Float16* KTp = KTb + ((size_t)(b * NH + h)) * NDH * NN;
  const _Float16* Vp  = Vb  + ((size_t)(b * NH + h)) * NN * NDH;

  v16h a0 = load_a_frag(Qp, NDH, t0, 0, lane);
  v16h a1 = load_a_frag(Qp, NDH, t0, 32, lane);

  v8f o0 = (v8f){}, o1 = (v8f){}, o2 = (v8f){}, o3 = (v8f){};
  float mrow[8], lrow[8];
#pragma unroll
  for (int r = 0; r < 8; ++r) { mrow[r] = -1e30f; lrow[r] = 0.f; }

  const float SCALE = 0.125f;            // DH^-0.5
  int ntiles = qseg / 16 + 2;            // visible 16-key tiles (pm + causal)
  int npairs = (ntiles + 1) / 2;

  for (int p = 0; p < npairs; ++p) {
    v8f s[2];
#pragma unroll
    for (int u = 0; u < 2; ++u) {
      int jt = 32 * p + 16 * u;          // logical key-tile base (0..)
      v16h bk0, bk1;
      if (jt == 0) {                     // persistent-memory keys
        int n = lane & 15, kb = 8 * (lane >> 4);
#pragma unroll
        for (int i = 0; i < 16; ++i) {
          int kd = kb + (i & 7) + ((i >> 3) << 4);
          bk0[i] = PMKT[((size_t)h * NDH + kd) * NPM + n];
          bk1[i] = PMKT[((size_t)h * NDH + 32 + kd) * NPM + n];
        }
      } else {
        int cb = jt - NPM;
        if (cb + 16 > NSEG) cb = NSEG - 16;     // clamp (tile fully masked)
        int col = w * NSEG + cb;
        int n = lane & 15, kb = 8 * (lane >> 4);
#pragma unroll
        for (int i = 0; i < 16; ++i) {
          int kd = kb + (i & 7) + ((i >> 3) << 4);
          bk0[i] = KTp[(size_t)kd * NN + col + n];
          bk1[i] = KTp[(size_t)(32 + kd) * NN + col + n];
        }
      }
      v8f z = (v8f){};
      z = wmma_f16(a0, bk0, z);
      z = wmma_f16(a1, bk1, z);
      s[u] = z;
    }

    int jc0 = 32 * p + (lane & 15);
    int jc1 = jc0 + 16;
#pragma unroll
    for (int r = 0; r < 8; ++r) {
      int i = qseg + r + 8 * (lane >> 4);      // query row within segment
      float x0 = s[0][r] * SCALE; if (jc0 > i + NPM) x0 = -1e30f;
      float x1 = s[1][r] * SCALE; if (jc1 > i + NPM) x1 = -1e30f;
      float mx = fmaxf(x0, x1);
#pragma unroll
      for (int d = 1; d < 16; d <<= 1) mx = fmaxf(mx, __shfl_xor(mx, d, 32));
      mx = fmaxf(mx, mrow[r]);
      float alpha = __expf(mrow[r] - mx);
      float p0 = __expf(x0 - mx);
      float p1 = __expf(x1 - mx);
      float rs = p0 + p1;
#pragma unroll
      for (int d = 1; d < 16; d <<= 1) rs += __shfl_xor(rs, d, 32);
      lrow[r] = lrow[r] * alpha + rs;
      mrow[r] = mx;
      o0[r] *= alpha; o1[r] *= alpha; o2[r] *= alpha; o3[r] *= alpha;
      int m = r + 8 * (lane >> 4);
      Plds[wave][m][lane & 15]        = (_Float16)p0;
      Plds[wave][m][16 + (lane & 15)] = (_Float16)p1;
    }

    // C-layout -> A-layout transpose via LDS (within-wave, explicit DS wait)
    asm volatile("s_wait_dscnt 0" ::: "memory");
    __builtin_amdgcn_wave_barrier();

    v16h pa;
    {
      int m  = lane & 15;
      int kk = 8 * (lane >> 4);
      v8h lo = *(const v8h*)&Plds[wave][m][kk];
      v8h hi = *(const v8h*)&Plds[wave][m][kk + 16];
#pragma unroll
      for (int i = 0; i < 8; ++i) { pa[i] = lo[i]; pa[i + 8] = hi[i]; }
    }

    // V B-fragments: lane -> dh column (lane&15); elements -> keys 8*(lane>>4)+{0..7,16..23}
    v16h vb0, vb1, vb2, vb3;
#pragma unroll
    for (int e = 0; e < 16; ++e) {
      int j = 32 * p + 8 * (lane >> 4) + (e & 7) + ((e >> 3) << 4);
      const _Float16* vrow;
      if (j < NPM) {
        vrow = PMV + ((size_t)h * NPM + j) * NDH;
      } else {
        int tk = j - NPM; if (tk >= NSEG) tk = NSEG - 1;  // clamped, P==0
        vrow = Vp + ((size_t)(w * NSEG + tk)) * NDH;
      }
      int n = lane & 15;
      vb0[e] = vrow[n];
      vb1[e] = vrow[16 + n];
      vb2[e] = vrow[32 + n];
      vb3[e] = vrow[48 + n];
    }
    o0 = wmma_f16(pa, vb0, o0);
    o1 = wmma_f16(pa, vb1, o1);
    o2 = wmma_f16(pa, vb2, o2);
    o3 = wmma_f16(pa, vb3, o3);
  }

  // Normalize and store to [B, N, H*DH] f16
#pragma unroll
  for (int r = 0; r < 8; ++r) {
    int m = r + 8 * (lane >> 4);
    int t = t0 + m;
    float inv = 1.0f / lrow[r];
    size_t ro = ((size_t)b * NN + t) * (NH * NDH) + h * NDH + (lane & 15);
    AttO[ro + 0]  = (_Float16)(o0[r] * inv);
    AttO[ro + 16] = (_Float16)(o1[r] * inv);
    AttO[ro + 32] = (_Float16)(o2[r] * inv);
    AttO[ro + 48] = (_Float16)(o3[r] * inv);
  }
}

// ---------------------------------------------------------------------------
// Output GEMM (8192x1024 @ 1024x1024) + bias, f32 output.
// ---------------------------------------------------------------------------
__global__ void out_gemm_kernel(const _Float16* __restrict__ A,
                                const _Float16* __restrict__ W,
                                const float* __restrict__ bias,
                                float* __restrict__ Out) {
  const int K = ND, Nout = ND;
  int lane = threadIdx.x & 31;
  int wid  = blockIdx.x * 8 + (threadIdx.x >> 5);
  const int WN = Nout / 64;            // 16
  int wm = wid / WN, wn = wid % WN;
  int m0 = wm * 32, n0 = wn * 64;

  v8f acc[2][4];
#pragma unroll
  for (int i = 0; i < 2; ++i)
#pragma unroll
    for (int j = 0; j < 4; ++j) acc[i][j] = (v8f){};

  for (int k = 0; k < K; k += 32) {
    v16h a0 = load_a_frag(A, K, m0, k, lane);
    v16h a1 = load_a_frag(A, K, m0 + 16, k, lane);
    v16h bf[4];
#pragma unroll
    for (int j = 0; j < 4; ++j) bf[j] = load_b_frag(W, Nout, k, n0 + 16 * j, lane);
#pragma unroll
    for (int j = 0; j < 4; ++j) {
      acc[0][j] = wmma_f16(a0, bf[j], acc[0][j]);
      acc[1][j] = wmma_f16(a1, bf[j], acc[1][j]);
    }
  }

#pragma unroll
  for (int i = 0; i < 2; ++i) {
#pragma unroll
    for (int j = 0; j < 4; ++j) {
      int n = n0 + 16 * j + (lane & 15);
      float bv = bias[n];
#pragma unroll
      for (int r = 0; r < 8; ++r) {
        int row = m0 + 16 * i + r + 8 * (lane >> 4);
        Out[(size_t)row * Nout + n] = acc[i][j][r] + bv;
      }
    }
  }
}

// ---------------------------------------------------------------------------
// Host-side launcher
// ---------------------------------------------------------------------------
extern "C" void kernel_launch(void* const* d_in, const int* in_sizes, int n_in,
                              void* d_out, int out_size, void* d_ws, size_t ws_size,
                              hipStream_t stream) {
  (void)in_sizes; (void)n_in; (void)out_size; (void)ws_size;

  const float* seq   = (const float*)d_in[0];  // [B,N,D]
  const float* gamma = (const float*)d_in[1];  // [D]
  const float* w_qkv = (const float*)d_in[2];  // [D, 3*H*DH]
  const float* w_out = (const float*)d_in[3];  // [H*DH, D]
  const float* b_out = (const float*)d_in[4];  // [D]
  const float* pmem  = (const float*)d_in[5];  // [2,H,PM,DH]
  float* out = (float*)d_out;                  // [B,N,D] f32

  // Workspace partition (bytes), all 256B-aligned
  char* ws = (char*)d_ws;
  size_t off = 0;
  _Float16* Xn   = (_Float16*)(ws + off); off += (size_t)NROWS * ND * 2;          // 16 MB
  _Float16* WqH  = (_Float16*)(ws + off); off += (size_t)ND * NQKV * 2;           // 6 MB
  _Float16* WoH  = (_Float16*)(ws + off); off += (size_t)ND * ND * 2;             // 2 MB
  _Float16* Qb   = (_Float16*)(ws + off); off += (size_t)NB * NH * NN * NDH * 2;  // 16 MB
  _Float16* KTb  = (_Float16*)(ws + off); off += (size_t)NB * NH * NDH * NN * 2 + 256;
  _Float16* Vb   = (_Float16*)(ws + off); off += (size_t)NB * NH * NN * NDH * 2 + 256;
  _Float16* PMKT = (_Float16*)(ws + off); off += (size_t)NH * NDH * NPM * 2;
  _Float16* PMV  = (_Float16*)(ws + off); off += (size_t)NH * NPM * NDH * 2;
  _Float16* AttO = (_Float16*)(ws + off); off += (size_t)NROWS * ND * 2;          // 16 MB
  float*    cosT = (float*)(ws + off);    off += (size_t)NN * NDH * 4;            // 1 MB
  float*    sinT = (float*)(ws + off);    off += (size_t)NN * NDH * 4;            // 1 MB

  // 1) Weight / pmem conversion + RoPE tables
  cvt_f32_to_f16<<<(ND * NQKV) / 256, 256, 0, stream>>>(w_qkv, WqH, ND * NQKV);
  cvt_f32_to_f16<<<(ND * ND) / 256, 256, 0, stream>>>(w_out, WoH, ND * ND);
  cvt_pmem<<<(NH * NPM * NDH) / 256, 256, 0, stream>>>(pmem, PMKT, PMV);
  rope_table_kernel<<<(NN * NDH) / 256, 256, 0, stream>>>(cosT, sinT);

  // 2) RMSNorm
  rmsnorm_kernel<<<NROWS, 256, 0, stream>>>(seq, gamma, Xn);

  // 3) QKV GEMM + RoPE + scatter (8192/32 * 3072/64 = 12288 waves)
  qkv_gemm_kernel<<<(NROWS / 32) * (NQKV / 64) / 8, 256, 0, stream>>>(
      Xn, WqH, cosT, sinT, Qb, KTb, Vb);

  // 4) Attention (8192 waves: one per 16 query rows per (b,h))
  attn_kernel<<<(NB * NH * NN / 16) / 8, 256, 0, stream>>>(Qb, KTb, Vb, PMKT, PMV, AttO);

  // 5) Output projection + bias (8192/32 * 1024/64 = 4096 waves)
  out_gemm_kernel<<<(NROWS / 32) * (ND / 64) / 8, 256, 0, stream>>>(AttO, WoH, b_out, out);
}
